// SparseMOE_26448408609193
// MI455X (gfx1250) — matrix operand order repaired and checked by reading, not compiled, not run on hardware
//
#include <hip/hip_runtime.h>
#include <stdint.h>

// ---------------------------------------------------------------------------
// Sparse MoE for MI455X (gfx1250, wave32, WMMA).
//   gate:   H = X @ W1^T + b1            (bf16 WMMA GEMM, 34 GFLOP)
//   router: logits = H @ W2^T + b2 -> softmax -> top2 -> renorm (VALU)
//   group:  deterministic counting-sort of (token, expert) pairs
//   expert: 8 sequential token-gathered bf16 WMMA GEMMs, out accumulated
//           in fixed expert order (write-then-add; no atomics, no memset).
// Roofline: ~300MB traffic -> 13us @ 23.3TB/s; 137 GFLOP bf16 -> ~25-30us.
// ---------------------------------------------------------------------------

typedef __bf16 bf16;
typedef __attribute__((ext_vector_type(16))) __bf16 bf16x16;
typedef __attribute__((ext_vector_type(8)))  __bf16 bf16x8;
typedef __attribute__((ext_vector_type(4)))  __bf16 bf16x4;
typedef __attribute__((ext_vector_type(8)))  float  f32x8;

#define T_TOK  32768
#define DIN    1024
#define DOUTD  1024
#define DH     512
#define NEXP   8

#define BM 128
#define BN 128
#define BK 32
#define LDSK 40   /* padded LDS row stride (elements) to dodge bank conflicts */

#ifndef USE_ASYNC_LDS
#define USE_ASYNC_LDS 1
#endif

// ---- 16B global -> LDS async copy (CDNA5 ASYNCcnt path) -------------------
__device__ __forceinline__ void async_cp16(const void* g, void* lds) {
#if USE_ASYNC_LDS
  asm volatile("global_load_async_to_lds_b128 %0, %1, off"
               :: "v"((unsigned)(uintptr_t)lds),
                  "v"((unsigned long long)(uintptr_t)g)
               : "memory");
#else
  *(uint4*)lds = *(const uint4*)g;
#endif
}
__device__ __forceinline__ void wait_async_4() {
#if USE_ASYNC_LDS
  asm volatile("s_wait_asynccnt 0x4" ::: "memory");
#endif
}
__device__ __forceinline__ void wait_async_0() {
#if USE_ASYNC_LDS
  asm volatile("s_wait_asynccnt 0x0" ::: "memory");
#endif
}

// each thread stages 32B of A and 32B of B (row r, half h) -> 4 async ops/wave
__device__ __forceinline__ void stage32(const bf16* ga, const bf16* gb,
                                        bf16* la, bf16* lb) {
  async_cp16(ga,     la);
  async_cp16(ga + 8, la + 8);
  async_cp16(gb,     lb);
  async_cp16(gb + 8, lb + 8);
}

// ---- wave-level 32x64 C tile, one K=32 step -------------------------------
// A frag (16x32 bf16): lane l<16 -> row M=l, K 0..7 & 16..23; l>=16 -> K 8..15 & 24..31
// B frag (32x16 bf16): lane l -> col N=l&15, K same as A frag (W[n][k] rows)
__device__ __forceinline__ void mma_step(const bf16* __restrict__ As,
                                         const bf16* __restrict__ Bs,
                                         int lane, int wm, int wn,
                                         f32x8 (&acc)[2][4]) {
  const int lm = lane & 15;
  const int kh = lane >> 4;
  bf16x16 a[2];
#pragma unroll
  for (int mt = 0; mt < 2; ++mt) {
    const bf16* p = As + (wm * 32 + mt * 16 + lm) * LDSK + kh * 8;
    bf16x8 lo = *(const bf16x8*)p;
    bf16x8 hi = *(const bf16x8*)(p + 16);
#pragma unroll
    for (int i = 0; i < 8; ++i) { a[mt][i] = lo[i]; a[mt][i + 8] = hi[i]; }
  }
#pragma unroll
  for (int nt = 0; nt < 4; ++nt) {
    const bf16* p = Bs + (wn * 64 + nt * 16 + lm) * LDSK + kh * 8;
    bf16x8 lo = *(const bf16x8*)p;
    bf16x8 hi = *(const bf16x8*)(p + 16);
    bf16x16 b;
#pragma unroll
    for (int i = 0; i < 8; ++i) { b[i] = lo[i]; b[i + 8] = hi[i]; }
#pragma unroll
    for (int mt = 0; mt < 2; ++mt)
      acc[mt][nt] = __builtin_amdgcn_wmma_f32_16x16x32_bf16(
          false, a[mt], false, b, (short)0, acc[mt][nt], false, false);
  }
}

// ---------------------------------------------------------------------------
// fp32 -> bf16 conversion (vector4)
// ---------------------------------------------------------------------------
__global__ void cvt_f32_bf16(const float* __restrict__ s, bf16* __restrict__ d,
                             int n4) {
  int i = blockIdx.x * blockDim.x + threadIdx.x;
  if (i >= n4) return;
  float4 v = ((const float4*)s)[i];
  bf16x4 o;
  o[0] = (bf16)v.x; o[1] = (bf16)v.y; o[2] = (bf16)v.z; o[3] = (bf16)v.w;
  ((bf16x4*)d)[i] = o;
}

// ---------------------------------------------------------------------------
// Gate layer-1 GEMM: H[T,512] = X[T,1024] @ W1[512,1024]^T + b1
// ---------------------------------------------------------------------------
__global__ __launch_bounds__(256) void gate_h_gemm(
    const bf16* __restrict__ X, const bf16* __restrict__ W1,
    const float* __restrict__ b1, bf16* __restrict__ H) {
  __shared__ bf16 As[2][BM * LDSK];
  __shared__ bf16 Bs[2][BN * LDSK];
  const int tid = threadIdx.x, lane = tid & 31, wv = tid >> 5;
  const int wm = wv >> 1, wn = wv & 1;
  const int m0 = blockIdx.x * BM, n0 = blockIdx.y * BN;
  const int r = tid & 127, h = (tid >> 7) & 1;

  const bf16* arow = X  + (size_t)(m0 + r) * DIN + h * 16;
  const bf16* brow = W1 + (size_t)(n0 + r) * DIN + h * 16;
  bf16* la[2] = { &As[0][r * LDSK + h * 16], &As[1][r * LDSK + h * 16] };
  bf16* lb[2] = { &Bs[0][r * LDSK + h * 16], &Bs[1][r * LDSK + h * 16] };

  f32x8 acc[2][4];
#pragma unroll
  for (int a = 0; a < 2; ++a)
#pragma unroll
    for (int b = 0; b < 4; ++b)
#pragma unroll
      for (int i = 0; i < 8; ++i) acc[a][b][i] = 0.f;

  const int NK = DIN / BK;
  stage32(arow, brow, la[0], lb[0]);
  int buf = 0;
  for (int k = 0; k < NK; ++k) {
    if (k + 1 < NK) {
      stage32(arow + (k + 1) * BK, brow + (k + 1) * BK, la[buf ^ 1], lb[buf ^ 1]);
      wait_async_4();
    } else {
      wait_async_0();
    }
    __syncthreads();
    mma_step(&As[buf][0], &Bs[buf][0], lane, wm, wn, acc);
    __syncthreads();
    buf ^= 1;
  }

#pragma unroll
  for (int mt = 0; mt < 2; ++mt)
#pragma unroll
    for (int nt = 0; nt < 4; ++nt) {
      int col = n0 + wn * 64 + nt * 16 + (lane & 15);
      float bias = b1[col];
      int rb = m0 + wm * 32 + mt * 16 + (lane >> 4) * 8;
#pragma unroll
      for (int j = 0; j < 8; ++j)
        H[(size_t)(rb + j) * DH + col] = (bf16)(acc[mt][nt][j] + bias);
    }
}

// ---------------------------------------------------------------------------
// Router: logits, softmax, top-2, renormalized combine weights cw[T,8]
// ---------------------------------------------------------------------------
__global__ __launch_bounds__(256) void router_kernel(
    const bf16* __restrict__ H, const float* __restrict__ w2,
    const float* __restrict__ b2, float* __restrict__ logits,
    float* __restrict__ cw) {
  __shared__ bf16  Hs[32 * DH];
  __shared__ float W2s[NEXP * DH];
  __shared__ float lg[32 * NEXP];
  const int tid = threadIdx.x;
  const int t0 = blockIdx.x * 32;

  for (int i = tid; i < NEXP * DH / 4; i += 256)
    ((float4*)W2s)[i] = ((const float4*)w2)[i];
  const uint4* hsrc = (const uint4*)(H + (size_t)t0 * DH);
  for (int i = tid; i < 32 * DH / 8; i += 256) ((uint4*)Hs)[i] = hsrc[i];
  __syncthreads();

  {  // 32 tokens x 8 experts == 256 threads, one logit each
    int tok = tid & 31, e = tid >> 5;
    float a = b2[e];
    const bf16*  hp = Hs + tok * DH;
    const float* wp = W2s + e * DH;
    for (int k = 0; k < DH; ++k) a += (float)hp[k] * wp[k];
    lg[tok * NEXP + e] = a;
  }
  __syncthreads();

  if (tid < 32) {
    int t = t0 + tid;
    float l[8];
#pragma unroll
    for (int e = 0; e < 8; ++e) {
      l[e] = lg[tid * 8 + e];
      logits[(size_t)t * 8 + e] = l[e];
    }
    float m = l[0];
#pragma unroll
    for (int e = 1; e < 8; ++e) m = fmaxf(m, l[e]);
    float p[8];
#pragma unroll
    for (int e = 0; e < 8; ++e) p[e] = __expf(l[e] - m);
    int i0 = 0;
#pragma unroll
    for (int e = 1; e < 8; ++e) if (p[e] > p[i0]) i0 = e;
    int i1 = (i0 == 0) ? 1 : 0;
#pragma unroll
    for (int e = 0; e < 8; ++e) if (e != i0 && p[e] > p[i1]) i1 = e;
    float den = p[i0] + p[i1];
    float c[8];
#pragma unroll
    for (int e = 0; e < 8; ++e) c[e] = 0.f;
    c[i0] = p[i0] / den;
    c[i1] = p[i1] / den;
    float4* co = (float4*)(cw + (size_t)t * 8);
    co[0] = make_float4(c[0], c[1], c[2], c[3]);
    co[1] = make_float4(c[4], c[5], c[6], c[7]);
  }
}

// ---------------------------------------------------------------------------
// Deterministic token grouping: count -> scan -> emit
// ---------------------------------------------------------------------------
__global__ void count_kernel(const float* __restrict__ cw,
                             unsigned* __restrict__ counts) {
  __shared__ unsigned cnt[NEXP];
  int tid = threadIdx.x;  // 128
  if (tid < NEXP) cnt[tid] = 0;
  __syncthreads();
  const float* row = cw + ((size_t)blockIdx.x * 128 + tid) * NEXP;
#pragma unroll
  for (int e = 0; e < NEXP; ++e)
    if (row[e] != 0.f) atomicAdd(&cnt[e], 1u);
  __syncthreads();
  if (tid < NEXP) counts[blockIdx.x * NEXP + tid] = cnt[tid];
}

__global__ void scan_kernel(const unsigned* __restrict__ counts,
                            unsigned* __restrict__ bbase,
                            unsigned* __restrict__ estart) {
  __shared__ unsigned c[256 * NEXP];
  __shared__ unsigned tot[NEXP];
  int tid = threadIdx.x;  // 256
  for (int i = tid; i < 256 * NEXP; i += 256) c[i] = counts[i];
  __syncthreads();
  if (tid < NEXP) {
    unsigned run = 0;
    for (int b = 0; b < 256; ++b) {
      unsigned v = c[b * NEXP + tid];
      c[b * NEXP + tid] = run;
      run += v;
    }
    tot[tid] = run;
  }
  __syncthreads();
  if (tid == 0) {
    unsigned run = 0;
    for (int e = 0; e < NEXP; ++e) { estart[e] = run; run += tot[e]; }
    estart[NEXP] = run;  // == 2*T
  }
  __syncthreads();
  for (int i = tid; i < 256 * NEXP; i += 256) bbase[i] = c[i];
}

__global__ void emit_kernel(const float* __restrict__ cw,
                            const unsigned* __restrict__ bbase,
                            const unsigned* __restrict__ estart,
                            unsigned* __restrict__ rtok,
                            float* __restrict__ rwgt) {
  __shared__ unsigned long long s[128];
  int tid = threadIdx.x;  // 128
  int t = blockIdx.x * 128 + tid;
  float w[NEXP];
  unsigned long long packed = 0;
#pragma unroll
  for (int e = 0; e < NEXP; ++e) {
    w[e] = cw[(size_t)t * NEXP + e];
    if (w[e] != 0.f) packed += 1ull << (8 * e);
  }
  s[tid] = packed;
  __syncthreads();
  for (int d = 1; d < 128; d <<= 1) {  // inclusive Hillis-Steele (bytes/expert)
    unsigned long long x = (tid >= d) ? s[tid - d] : 0ull;
    __syncthreads();
    s[tid] += x;
    __syncthreads();
  }
  unsigned long long incl = s[tid];
  bool first = true;
#pragma unroll
  for (int e = 0; e < NEXP; ++e)
    if (w[e] != 0.f) {
      unsigned local = (unsigned)((incl >> (8 * e)) & 0xFF) - 1u;  // exclusive
      unsigned pos = estart[e] + bbase[blockIdx.x * NEXP + e] + local;
      rtok[pos] = (unsigned)t | (first ? 0x80000000u : 0u);
      rwgt[pos] = w[e];
      first = false;
    }
}

// ---------------------------------------------------------------------------
// Expert GEMM (token-gathered), launched once per expert in ascending order.
// First contribution writes out, second accumulates -> deterministic, no memset.
// ---------------------------------------------------------------------------
__global__ __launch_bounds__(256) void expert_gemm(
    const bf16* __restrict__ X, const bf16* __restrict__ We,
    const float* __restrict__ be, const unsigned* __restrict__ rtok,
    const float* __restrict__ rwgt, const unsigned* __restrict__ estart,
    int e, float* __restrict__ out) {
  const int rs = (int)estart[e], re = (int)estart[e + 1];
  if ((int)blockIdx.x * BM >= re - rs) return;  // uniform early exit

  __shared__ bf16 As[2][BM * LDSK];
  __shared__ bf16 Bs[2][BN * LDSK];
  const int tid = threadIdx.x, lane = tid & 31, wv = tid >> 5;
  const int wm = wv >> 1, wn = wv & 1;
  const int n0 = blockIdx.y * BN;
  const int r = tid & 127, h = (tid >> 7) & 1;

  const int gi = rs + blockIdx.x * BM + r;
  const int tok = (gi < re) ? (int)(rtok[gi] & 0x7fffffffu) : 0;
  const bf16* arow = X + (size_t)tok * DIN + h * 16;
  const bf16* brow = We + ((size_t)e * DOUTD + n0 + r) * DIN + h * 16;
  bf16* la[2] = { &As[0][r * LDSK + h * 16], &As[1][r * LDSK + h * 16] };
  bf16* lb[2] = { &Bs[0][r * LDSK + h * 16], &Bs[1][r * LDSK + h * 16] };

  f32x8 acc[2][4];
#pragma unroll
  for (int a = 0; a < 2; ++a)
#pragma unroll
    for (int b = 0; b < 4; ++b)
#pragma unroll
      for (int i = 0; i < 8; ++i) acc[a][b][i] = 0.f;

  const int NK = DIN / BK;
  stage32(arow, brow, la[0], lb[0]);
  int buf = 0;
  for (int k = 0; k < NK; ++k) {
    if (k + 1 < NK) {
      stage32(arow + (k + 1) * BK, brow + (k + 1) * BK, la[buf ^ 1], lb[buf ^ 1]);
      wait_async_4();
    } else {
      wait_async_0();
    }
    __syncthreads();
    mma_step(&As[buf][0], &Bs[buf][0], lane, wm, wn, acc);
    __syncthreads();
    buf ^= 1;
  }

#pragma unroll
  for (int mt = 0; mt < 2; ++mt)
#pragma unroll
    for (int nt = 0; nt < 4; ++nt) {
      int col = n0 + wn * 64 + nt * 16 + (lane & 15);
      float bias = be[(size_t)e * DOUTD + col];
      int rb = blockIdx.x * BM + wm * 32 + mt * 16 + (lane >> 4) * 8;
#pragma unroll
      for (int j = 0; j < 8; ++j) {
        int g2 = rs + rb + j;
        if (g2 < re) {
          unsigned rt = rtok[g2];
          int tk = (int)(rt & 0x7fffffffu);
          float val = rwgt[g2] * (acc[mt][nt][j] + bias);
          float* o = out + (size_t)tk * DOUTD + col;
          if (rt >> 31) *o = val; else *o += val;
        }
      }
    }
}

// ---------------------------------------------------------------------------
extern "C" void kernel_launch(void* const* d_in, const int* in_sizes, int n_in,
                              void* d_out, int out_size, void* d_ws,
                              size_t ws_size, hipStream_t stream) {
  (void)in_sizes; (void)n_in; (void)out_size; (void)ws_size;
  const float* x   = (const float*)d_in[0];
  const float* gw1 = (const float*)d_in[1];
  const float* gb1 = (const float*)d_in[2];
  const float* gw2 = (const float*)d_in[3];
  const float* gb2 = (const float*)d_in[4];
  const float* ew  = (const float*)d_in[5];
  const float* eb  = (const float*)d_in[6];
  float* out    = (float*)d_out;                     // [T, D_OUT]
  float* logits = out + (size_t)T_TOK * DOUTD;       // [T, E]

  char* ws = (char*)d_ws;
  size_t off = 0;
  auto take = [&](size_t bytes) -> void* {
    void* p = ws + off;
    off = (off + bytes + 255) & ~(size_t)255;
    return p;
  };
  bf16*     Xbf    = (bf16*)take((size_t)T_TOK * DIN * 2);          // 64 MB
  bf16*     Hbf    = (bf16*)take((size_t)T_TOK * DH * 2);           // 32 MB
  bf16*     W1bf   = (bf16*)take((size_t)DH * DIN * 2);             //  1 MB
  bf16*     Webf   = (bf16*)take((size_t)NEXP * DOUTD * DIN * 2);   // 16 MB
  float*    cw     = (float*)take((size_t)T_TOK * NEXP * 4);        //  1 MB
  unsigned* counts = (unsigned*)take(256 * NEXP * 4);
  unsigned* bbase  = (unsigned*)take(256 * NEXP * 4);
  unsigned* estart = (unsigned*)take(16 * 4);
  unsigned* rtok   = (unsigned*)take((size_t)2 * T_TOK * 4);
  float*    rwgt   = (float*)take((size_t)2 * T_TOK * 4);

  // 1) fp32 -> bf16 conversions
  {
    int n4 = T_TOK * DIN / 4;
    cvt_f32_bf16<<<(n4 + 255) / 256, 256, 0, stream>>>(x, Xbf, n4);
  }
  {
    int n4 = DH * DIN / 4;
    cvt_f32_bf16<<<(n4 + 255) / 256, 256, 0, stream>>>(gw1, W1bf, n4);
  }
  {
    int n4 = NEXP * DOUTD * DIN / 4;
    cvt_f32_bf16<<<(n4 + 255) / 256, 256, 0, stream>>>(ew, Webf, n4);
  }

  // 2) gate layer-1 GEMM (WMMA bf16)
  gate_h_gemm<<<dim3(T_TOK / BM, DH / BN), 256, 0, stream>>>(Xbf, W1bf, gb1, Hbf);

  // 3) router (logits + softmax + top-2 + renorm)
  router_kernel<<<T_TOK / 32, 256, 0, stream>>>(Hbf, gw2, gb2, logits, cw);

  // 4) deterministic grouping
  count_kernel<<<256, 128, 0, stream>>>(cw, counts);
  scan_kernel<<<1, 256, 0, stream>>>(counts, bbase, estart);
  emit_kernel<<<256, 128, 0, stream>>>(cw, bbase, estart, rtok, rwgt);

  // 5) expert GEMMs, sequential in expert order (deterministic accumulation)
  for (int e = 0; e < NEXP; ++e)
    expert_gemm<<<dim3(T_TOK / BM, DOUTD / BN), 256, 0, stream>>>(
        Xbf, Webf, eb, rtok, rwgt, estart, e, out);
}
